// GCN_layer_72688026518114
// MI455X (gfx1250) — compile-verified, matrix operand order checked by reading, not run on hardware
//
#include <hip/hip_runtime.h>

// ---------------- CDNA5 WMMA types ----------------
typedef __attribute__((ext_vector_type(16))) __bf16 v16bf;
typedef __attribute__((ext_vector_type(8)))  __bf16 v8bf;
typedef __attribute__((ext_vector_type(4)))  __bf16 v4bf;
typedef __attribute__((ext_vector_type(8)))  float  v8f;

#define NGR 64
#define DIM 512
#define EPSN 1e-5f

// GEMM tiling
#define BM 128
#define BN 64
#define BK 32
#define AS 40   // LDS row stride (elements): 80B rows -> 16B aligned, 20-bank step (conflict-free over 16 rows)

static __device__ __forceinline__ v16bf frag_cat(v8bf a, v8bf b) {
    return __builtin_shufflevector(a, b, 0,1,2,3,4,5,6,7,8,9,10,11,12,13,14,15);
}

// ---------------- small prep kernels ----------------
__global__ void k_init(float* __restrict__ deg, float* __restrict__ gsum,
                       float* __restrict__ gvar, float* __restrict__ gcnt, int N) {
    int i = blockIdx.x * blockDim.x + threadIdx.x;
    if (i < N) deg[i] = 1.0f;                 // self-loop weight
    if (i < NGR * DIM) { gsum[i] = 0.0f; gvar[i] = 0.0f; }
    if (i < NGR) gcnt[i] = 0.0f;
}

__global__ void k_deg(const int* __restrict__ ei, const float* __restrict__ ea,
                      float* __restrict__ deg, int E) {
    int e = blockIdx.x * blockDim.x + threadIdx.x;
    if (e < E) atomicAdd(&deg[ei[E + e]], ea[e]);   // weighted in-degree at col
}

__global__ void k_dis(const float* __restrict__ deg, float* __restrict__ dis,
                      const int* __restrict__ batch, float* __restrict__ gcnt, int N) {
    int i = blockIdx.x * blockDim.x + threadIdx.x;
    if (i < N) {
        float d = deg[i];
        dis[i] = (d > 0.0f) ? rsqrtf(d) : 0.0f;
        atomicAdd(&gcnt[batch[i]], 1.0f);
    }
}

// ---- one-time: W (k-major) -> bf16 hi/lo planes of plain W^T (n-major, k contiguous) ----
__global__ __launch_bounds__(256) void k_prep(const float* __restrict__ W,
                                              __bf16* __restrict__ WtH,
                                              __bf16* __restrict__ WtL) {
    int id = blockIdx.x * 256 + threadIdx.x;      // 0 .. 512*512-1
    int k = id >> 9;
    int n = id & (DIM - 1);
    float  x  = W[id];
    __bf16 hi = (__bf16)x;
    __bf16 lo = (__bf16)(x - (float)hi);
    WtH[(size_t)n * DIM + k] = hi;
    WtL[(size_t)n * DIM + k] = lo;
}

// ---------------- split-bf16 WMMA GEMM:  y = X @ W ;  agg = y * dis^2 (self loop) ----
__global__ __launch_bounds__(256) void k_gemm(const float* __restrict__ X,
                                              const __bf16* __restrict__ WtH,
                                              const __bf16* __restrict__ WtL,
                                              const float* __restrict__ dis,
                                              float* __restrict__ y,
                                              float* __restrict__ agg, int N) {
    __shared__ alignas(16) __bf16 Ah[BM * AS];
    __shared__ alignas(16) __bf16 Al[BM * AS];

    const int t     = threadIdx.x;
    const int lane  = t & 31;
    const int wv    = t >> 5;            // 8 waves, each owns a 16-row strip
    const int mBase = blockIdx.y * BM;
    const int nBase = blockIdx.x * BN;
    const int nIdx  = lane & 15;
    const int half  = lane >> 4;

    // per-lane B fragment row bases (plain n-major W^T, L2-resident)
    const __bf16* bH[4];
    const __bf16* bL[4];
    #pragma unroll
    for (int nt = 0; nt < 4; ++nt) {
        size_t roff = (size_t)(nBase + nt * 16 + nIdx) * DIM;
        bH[nt] = WtH + roff;
        bL[nt] = WtL + roff;
    }

    v8f acc[4];
    #pragma unroll
    for (int q = 0; q < 4; ++q)
        #pragma unroll
        for (int r = 0; r < 8; ++r) acc[q][r] = 0.0f;

    for (int kk = 0; kk < DIM; kk += BK) {
        // ---- stage A tile (128x32 fp32 -> bf16 hi/lo split), vectorized b64 stores ----
        #pragma unroll
        for (int it = 0; it < 4; ++it) {
            int r  = it * 32 + (t >> 3);
            int c  = (t & 7) * 4;
            int gr = mBase + r;
            float4 v4 = make_float4(0.f, 0.f, 0.f, 0.f);
            if (gr < N) {
                v4 = *reinterpret_cast<const float4*>(X + (size_t)gr * DIM + kk + c);
                if (kk + BK < DIM)   // prefetch next K-slice (global_prefetch_b8)
                    __builtin_prefetch(X + (size_t)gr * DIM + kk + BK + c, 0, 3);
            }
            float xs[4] = {v4.x, v4.y, v4.z, v4.w};
            v4bf hi4, lo4;
            #pragma unroll
            for (int j = 0; j < 4; ++j) {
                float  xv = xs[j];
                __bf16 hb = (__bf16)xv;
                hi4[j] = hb;
                lo4[j] = (__bf16)(xv - (float)hb);
            }
            *reinterpret_cast<v4bf*>(Ah + r * AS + c) = hi4;   // 8B-aligned ds_store_b64
            *reinterpret_cast<v4bf*>(Al + r * AS + c) = lo4;
        }
        __syncthreads();

        // ---- A fragments: two contiguous 16B runs per lane (ISA 7.12.2) ----
        const __bf16* arH = Ah + (wv * 16 + nIdx) * AS;
        const __bf16* arL = Al + (wv * 16 + nIdx) * AS;
        v16bf ah = frag_cat(*reinterpret_cast<const v8bf*>(arH + half * 8),
                            *reinterpret_cast<const v8bf*>(arH + 16 + half * 8));
        v16bf al = frag_cat(*reinterpret_cast<const v8bf*>(arL + half * 8),
                            *reinterpret_cast<const v8bf*>(arL + 16 + half * 8));
        __syncthreads();   // LDS reads done before next fill

        // ---- B fragments straight from L2-resident plain n-major W^T ----
        #pragma unroll
        for (int nt = 0; nt < 4; ++nt) {
            const __bf16* rh = bH[nt] + kk;
            const __bf16* rl = bL[nt] + kk;
            v16bf bh = frag_cat(*reinterpret_cast<const v8bf*>(rh + half * 8),
                                *reinterpret_cast<const v8bf*>(rh + 16 + half * 8));
            v16bf bl = frag_cat(*reinterpret_cast<const v8bf*>(rl + half * 8),
                                *reinterpret_cast<const v8bf*>(rl + 16 + half * 8));
            // split accumulation: hi*hi + hi*lo + lo*hi (fp32 accumulate)
            acc[nt] = __builtin_amdgcn_wmma_f32_16x16x32_bf16(false, ah, false, bh,
                                                              (short)0, acc[nt], false, false);
            acc[nt] = __builtin_amdgcn_wmma_f32_16x16x32_bf16(false, ah, false, bl,
                                                              (short)0, acc[nt], false, false);
            acc[nt] = __builtin_amdgcn_wmma_f32_16x16x32_bf16(false, al, false, bh,
                                                              (short)0, acc[nt], false, false);
        }
    }

    // ---- epilogue: C/D layout — VGPR v holds row (v + 8*half), col = lane&15 ----
    #pragma unroll
    for (int nt = 0; nt < 4; ++nt) {
        #pragma unroll
        for (int vi = 0; vi < 8; ++vi) {
            int row = mBase + wv * 16 + vi + 8 * half;
            if (row < N) {
                int col = nBase + nt * 16 + nIdx;
                size_t off = (size_t)row * DIM + col;
                float val = acc[nt][vi];
                y[off] = val;
                float ds = dis[row];
                agg[off] = val * ds * ds;   // self-loop message pre-seeded
            }
        }
    }
}

// ---------------- edge scatter: agg[col] += y[row] * (dis[row]*w*dis[col]) --------
__global__ __launch_bounds__(256) void k_scatter(const int* __restrict__ ei,
                                                 const float* __restrict__ ea,
                                                 const float* __restrict__ dis,
                                                 const float* __restrict__ y,
                                                 float* __restrict__ agg, int E) {
    int e = blockIdx.x;
    int row = ei[e];
    int col = ei[E + e];
    float coef = dis[row] * ea[e] * dis[col];
    const float* ys = y   + (size_t)row * DIM;
    float*       as = agg + (size_t)col * DIM;
    int t = threadIdx.x;
    atomicAdd(&as[t],       ys[t]       * coef);
    atomicAdd(&as[t + 256], ys[t + 256] * coef);
}

// ------- h = node + conv + b (in place in agg) + per-graph sums (batch is sorted) --
__global__ __launch_bounds__(256) void k_hsum(const float* __restrict__ node,
                                              float* __restrict__ hbuf,
                                              const float* __restrict__ b,
                                              const int* __restrict__ batch,
                                              float* __restrict__ gsum, int N) {
    int t = threadIdx.x;
    int d0 = t * 2;
    int i0 = blockIdx.x * 64;
    int iend = (i0 + 64 < N) ? (i0 + 64) : N;
    float b0 = b[d0], b1 = b[d0 + 1];
    float s0 = 0.f, s1 = 0.f;
    int cur = batch[i0];
    for (int i = i0; i < iend; ++i) {
        int g = batch[i];
        if (g != cur) {
            atomicAdd(&gsum[cur * DIM + d0],     s0);
            atomicAdd(&gsum[cur * DIM + d0 + 1], s1);
            s0 = s1 = 0.f; cur = g;
        }
        size_t off = (size_t)i * DIM + d0;
        float2 nv = *reinterpret_cast<const float2*>(node + off);
        float2 av = *reinterpret_cast<const float2*>(hbuf + off);
        float h0 = nv.x + av.x + b0;
        float h1 = nv.y + av.y + b1;
        *reinterpret_cast<float2*>(hbuf + off) = make_float2(h0, h1);
        s0 += h0; s1 += h1;
    }
    atomicAdd(&gsum[cur * DIM + d0],     s0);
    atomicAdd(&gsum[cur * DIM + d0 + 1], s1);
}

// ------- centered = h - mean*alpha (in place), per-graph var sums ------------------
__global__ __launch_bounds__(256) void k_cvar(float* __restrict__ hbuf,
                                              const float* __restrict__ gsum,
                                              const float* __restrict__ gcnt,
                                              const float* __restrict__ alpha,
                                              const int* __restrict__ batch,
                                              float* __restrict__ gvar, int N) {
    int t = threadIdx.x;
    int d0 = t * 2;
    int i0 = blockIdx.x * 64;
    int iend = (i0 + 64 < N) ? (i0 + 64) : N;
    float a0 = alpha[d0], a1 = alpha[d0 + 1];
    int cur = batch[i0];
    float invc = 1.0f / fmaxf(gcnt[cur], 1.0f);
    float m0 = gsum[cur * DIM + d0] * invc;
    float m1 = gsum[cur * DIM + d0 + 1] * invc;
    float s0 = 0.f, s1 = 0.f;
    for (int i = i0; i < iend; ++i) {
        int g = batch[i];
        if (g != cur) {
            atomicAdd(&gvar[cur * DIM + d0],     s0);
            atomicAdd(&gvar[cur * DIM + d0 + 1], s1);
            s0 = s1 = 0.f; cur = g;
            invc = 1.0f / fmaxf(gcnt[cur], 1.0f);
            m0 = gsum[cur * DIM + d0] * invc;
            m1 = gsum[cur * DIM + d0 + 1] * invc;
        }
        size_t off = (size_t)i * DIM + d0;
        float2 hv = *reinterpret_cast<const float2*>(hbuf + off);
        float c0 = hv.x - m0 * a0;
        float c1 = hv.y - m1 * a1;
        *reinterpret_cast<float2*>(hbuf + off) = make_float2(c0, c1);
        s0 += c0 * c0; s1 += c1 * c1;
    }
    atomicAdd(&gvar[cur * DIM + d0],     s0);
    atomicAdd(&gvar[cur * DIM + d0 + 1], s1);
}

// ------- out = relu(w * centered * rsqrt(var/count + eps) + bias) ------------------
__global__ __launch_bounds__(256) void k_out(const float* __restrict__ cbuf,
                                             const float* __restrict__ gvar,
                                             const float* __restrict__ gcnt,
                                             const float* __restrict__ wgt,
                                             const float* __restrict__ bias,
                                             const int* __restrict__ batch,
                                             float* __restrict__ out, int N) {
    long long id = (long long)blockIdx.x * 256 + threadIdx.x;   // pair index
    long long total = (long long)N * (DIM / 2);
    if (id >= total) return;
    int i  = (int)(id / (DIM / 2));
    int d0 = (int)(id % (DIM / 2)) * 2;
    int g  = batch[i];
    float invc = 1.0f / fmaxf(gcnt[g], 1.0f);
    size_t off = (size_t)i * DIM + d0;
    float2 c = *reinterpret_cast<const float2*>(cbuf + off);
    float is0 = rsqrtf(gvar[g * DIM + d0] * invc + EPSN);
    float is1 = rsqrtf(gvar[g * DIM + d0 + 1] * invc + EPSN);
    float o0 = fmaxf(wgt[d0] * c.x * is0 + bias[d0], 0.0f);
    float o1 = fmaxf(wgt[d0 + 1] * c.y * is1 + bias[d0 + 1], 0.0f);
    *reinterpret_cast<float2*>(out + off) = make_float2(o0, o1);
}

// ---------------- launch ----------------
extern "C" void kernel_launch(void* const* d_in, const int* in_sizes, int n_in,
                              void* d_out, int out_size, void* d_ws, size_t ws_size,
                              hipStream_t stream) {
    const float* node  = (const float*)d_in[0];
    const float* ea    = (const float*)d_in[1];
    const float* W     = (const float*)d_in[2];
    const float* b     = (const float*)d_in[3];
    const float* gw    = (const float*)d_in[4];
    const float* gb    = (const float*)d_in[5];
    const float* galp  = (const float*)d_in[6];
    const int*   ei    = (const int*)d_in[7];
    const int*   batch = (const int*)d_in[8];

    const int E = in_sizes[1];   // 160000
    const int N = in_sizes[8];   // 50000

    float* out = (float*)d_out;

    // workspace layout
    char* ws = (char*)d_ws;
    size_t off = 0;
    auto wsAlloc = [&](size_t bytes) -> void* {
        void* p = ws + off;
        off += (bytes + 255) & ~(size_t)255;
        return p;
    };
    float*  y    = (float*)wsAlloc((size_t)N * DIM * sizeof(float));
    float*  agg  = (float*)wsAlloc((size_t)N * DIM * sizeof(float));
    __bf16* WtH  = (__bf16*)wsAlloc((size_t)DIM * DIM * sizeof(__bf16));
    __bf16* WtL  = (__bf16*)wsAlloc((size_t)DIM * DIM * sizeof(__bf16));
    float*  deg  = (float*)wsAlloc((size_t)N * sizeof(float));
    float*  dis  = (float*)wsAlloc((size_t)N * sizeof(float));
    float*  gsum = (float*)wsAlloc((size_t)NGR * DIM * sizeof(float));
    float*  gvar = (float*)wsAlloc((size_t)NGR * DIM * sizeof(float));
    float*  gcnt = (float*)wsAlloc((size_t)NGR * sizeof(float));

    int initN = (N > NGR * DIM) ? N : NGR * DIM;
    k_init<<<(initN + 255) / 256, 256, 0, stream>>>(deg, gsum, gvar, gcnt, N);
    k_deg<<<(E + 255) / 256, 256, 0, stream>>>(ei, ea, deg, E);
    k_dis<<<(N + 255) / 256, 256, 0, stream>>>(deg, dis, batch, gcnt, N);
    k_prep<<<(DIM * DIM) / 256, 256, 0, stream>>>(W, WtH, WtL);

    dim3 gGrid(DIM / BN, (N + BM - 1) / BM);   // 8 x 391
    k_gemm<<<gGrid, 256, 0, stream>>>(node, WtH, WtL, dis, y, agg, N);

    k_scatter<<<E, 256, 0, stream>>>(ei, ea, dis, y, agg, E);

    int nb64 = (N + 63) / 64;
    k_hsum<<<nb64, 256, 0, stream>>>(node, agg, b, batch, gsum, N);
    k_cvar<<<nb64, 256, 0, stream>>>(agg, gsum, gcnt, galp, batch, gvar, N);

    long long total = (long long)N * (DIM / 2);
    k_out<<<(int)((total + 255) / 256), 256, 0, stream>>>(agg, gvar, gcnt, gw, gb, batch, out, N);
}
